// FlashAttention2_57964878626974
// MI455X (gfx1250) — matrix operand order repaired and checked by reading, not compiled, not run on hardware
//
#include <hip/hip_runtime.h>
#include <hip/hip_bf16.h>

typedef __attribute__((ext_vector_type(16))) _Float16 v16h;
typedef __attribute__((ext_vector_type(8)))  float    v8f;

#define SEQ_LEN 8192
#define HEAD_D  128
#define BN      64      // keys per iteration (4 x 16 S-tiles)
#define WAVES   8       // wave32s per workgroup
#define BM      (WAVES * 16)

union AFrag {           // 16x32 f16 fragment (A or B operand), 8 VGPRs
    v16h v;
    uint4 u[2];
    _Float16 h[16];
};

__device__ __forceinline__ v8f wmma_f16(const v16h& a, const v16h& b, const v8f& c) {
    return __builtin_amdgcn_wmma_f32_16x16x32_f16(false, a, false, b, (short)0, c,
                                                  false, false);
}

// Async DMA of one contiguous 16 KB K tile (64 rows x 128 cols, row-major both in
// global and LDS) via GLOBAL_LOAD_ASYNC_TO_LDS_B128 (ASYNCcnt).
__device__ __forceinline__ void stage_k_async(const _Float16* __restrict__ K,
                                              int kb, _Float16* ldsdst, int tid) {
    const unsigned ldsbase = (unsigned)(uintptr_t)ldsdst;   // low 32 bits = LDS addr
    const _Float16* kg = K + (size_t)kb * HEAD_D;
#pragma unroll
    for (int c = 0; c < 4; ++c) {
        const int chunk = c * 256 + tid;                    // 1024 x 16B chunks
        const unsigned loff = ldsbase + chunk * 16;
        const unsigned long long gaddr =
            (unsigned long long)(uintptr_t)(kg + chunk * 8);
        asm volatile("global_load_async_to_lds_b128 %0, %1, off"
                     :: "v"(loff), "v"(gaddr) : "memory");
    }
}

__global__ __launch_bounds__(256) void
fa2_fwd_wmma_gfx1250(const _Float16* __restrict__ Q,
                     const _Float16* __restrict__ K,
                     const _Float16* __restrict__ V,
                     _Float16* __restrict__ Out,
                     _Float16* __restrict__ Lse)
{
    __shared__ alignas(16) _Float16 Ks[2][BN][HEAD_D];     // 2 x 16 KB, row-major
    __shared__ alignas(16) _Float16 Vts[2][HEAD_D][BN];    // 2 x 16 KB, transposed
    __shared__ alignas(16) _Float16 Ps[WAVES][16][BN];     // 16 KB, per-wave P

    const int tid  = threadIdx.x;
    const int wave = tid >> 5;
    const int lane = tid & 31;
    const int half = lane >> 4;
    const int l16  = lane & 15;

    const int qbase = blockIdx.x * BM + wave * 16;

    // ---- Q tile -> A-layout fragments (4 chunks of K=32 along d), lives in VGPRs
    AFrag aq[4];
    {
        const _Float16* qrow = Q + (size_t)(qbase + l16) * HEAD_D;
        const int b1 = half * 8;
#pragma unroll
        for (int c = 0; c < 4; ++c) {
            aq[c].u[0] = *(const uint4*)(qrow + c * 32 + b1);
            aq[c].u[1] = *(const uint4*)(qrow + c * 32 + b1 + 16);
        }
    }

    v8f  o_acc[8];
    float mrow[8], lrow[8];
    v8f  vzero = {};
#pragma unroll
    for (int t = 0; t < 8; ++t) o_acc[t] = vzero;
#pragma unroll
    for (int r = 0; r < 8; ++r) { mrow[r] = -1e30f; lrow[r] = 0.0f; }

    // V staging geometry: thread -> (row = tid/4, 32-col chunk = (tid%4)*32)
    const int vr  = tid >> 2;
    const int vc0 = (tid & 3) * 32;

    // ---- prologue: stage buffer 0
    stage_k_async(K, 0, &Ks[0][0][0], tid);
    {
        const uint4* vg = (const uint4*)(V + (size_t)vr * HEAD_D + vc0);
        uint4 vreg[4];
#pragma unroll
        for (int c = 0; c < 4; ++c) vreg[c] = vg[c];
#pragma unroll
        for (int c = 0; c < 4; ++c) {
            const _Float16* vh = (const _Float16*)&vreg[c];
#pragma unroll
            for (int j = 0; j < 8; ++j) Vts[0][vc0 + c * 8 + j][vr] = vh[j];
        }
    }
    asm volatile("s_wait_asynccnt 0x0" ::: "memory");
    __syncthreads();

    for (int kb = 0; kb < SEQ_LEN; kb += BN) {
        const int cur = (kb >> 6) & 1;
        const int nxt = cur ^ 1;
        const bool have_next = (kb + BN) < SEQ_LEN;

        // ---- kick off next tile: K via async DMA, V into registers
        uint4 vreg[4];
        if (have_next) {
            stage_k_async(K, kb + BN, &Ks[nxt][0][0], tid);
            const uint4* vg =
                (const uint4*)(V + (size_t)(kb + BN + vr) * HEAD_D + vc0);
#pragma unroll
            for (int c = 0; c < 4; ++c) vreg[c] = vg[c];
            if (kb + 2 * BN < SEQ_LEN)   // pull tile+2 toward L2
                __builtin_prefetch(V + (size_t)(kb + 2 * BN + vr) * HEAD_D + vc0, 0, 0);
        }

        // ---- S = Q K^T : 4 key-tiles x 4 d-chunks = 16 WMMAs
        v8f s[4];
#pragma unroll
        for (int kt = 0; kt < 4; ++kt) {
            v8f c = vzero;
            const _Float16* krow = &Ks[cur][kt * 16 + l16][0];
            const int b1 = half * 8;
#pragma unroll
            for (int cc = 0; cc < 4; ++cc) {
                AFrag bk;
                bk.u[0] = *(const uint4*)(krow + cc * 32 + b1);
                bk.u[1] = *(const uint4*)(krow + cc * 32 + b1 + 16);
                c = wmma_f16(aq[cc].v, bk.v, c);
            }
            s[kt] = c;
        }

        // ---- online softmax (C layout: row = r + 8*half, col = l16 + 16*kt)
        float pscale[8];
#pragma unroll
        for (int r = 0; r < 8; ++r) {
            float smax = fmaxf(fmaxf(s[0][r], s[1][r]), fmaxf(s[2][r], s[3][r]));
#pragma unroll
            for (int m = 1; m < 16; m <<= 1)
                smax = fmaxf(smax, __shfl_xor(smax, m, 16));
            float mnew = fmaxf(mrow[r], smax);
            float sc = __expf(mrow[r] - mnew);
            float rs = 0.0f;
#pragma unroll
            for (int kt = 0; kt < 4; ++kt) {
                float p = __expf(s[kt][r] - mnew);
                s[kt][r] = p;
                rs += p;
            }
#pragma unroll
            for (int m = 1; m < 16; m <<= 1)
                rs += __shfl_xor(rs, m, 16);
            lrow[r] = lrow[r] * sc + rs;
            mrow[r] = mnew;
            pscale[r] = sc;
        }
#pragma unroll
        for (int t = 0; t < 8; ++t)
#pragma unroll
            for (int r = 0; r < 8; ++r)
                o_acc[t][r] *= pscale[r];

        // ---- retire next-V registers into transposed LDS (latency now hidden)
        if (have_next) {
#pragma unroll
            for (int c = 0; c < 4; ++c) {
                const _Float16* vh = (const _Float16*)&vreg[c];
#pragma unroll
                for (int j = 0; j < 8; ++j) Vts[nxt][vc0 + c * 8 + j][vr] = vh[j];
            }
        }

        // ---- P (C layout f32) -> per-wave LDS row-major f16 (same-wave, in-order)
        {
            const int row = half * 8;
#pragma unroll
            for (int r = 0; r < 8; ++r)
#pragma unroll
                for (int kt = 0; kt < 4; ++kt)
                    Ps[wave][row + r][kt * 16 + l16] = (_Float16)s[kt][r];
        }
        // ---- reload P as two A-layout fragments (K = keys, 2 x 32)
        AFrag ap[2];
        {
            const _Float16* prow = &Ps[wave][l16][0];
            const int b1 = half * 8;
#pragma unroll
            for (int pc = 0; pc < 2; ++pc) {
                ap[pc].u[0] = *(const uint4*)(prow + pc * 32 + b1);
                ap[pc].u[1] = *(const uint4*)(prow + pc * 32 + b1 + 16);
            }
        }

        // ---- O += P V : 8 d-tiles x 2 key-chunks = 16 WMMAs
#pragma unroll
        for (int t = 0; t < 8; ++t) {
            const _Float16* vtr = &Vts[cur][t * 16 + l16][0];
            const int b1 = half * 8;
#pragma unroll
            for (int pc = 0; pc < 2; ++pc) {
                AFrag bv;
                bv.u[0] = *(const uint4*)(vtr + pc * 32 + b1);
                bv.u[1] = *(const uint4*)(vtr + pc * 32 + b1 + 16);
                o_acc[t] = wmma_f16(ap[pc].v, bv.v, o_acc[t]);
            }
        }

        // ---- close the pipeline stage: async K landed + everyone done with `cur`
        asm volatile("s_wait_asynccnt 0x0" ::: "memory");
        __syncthreads();
    }

    // ---- epilogue: O = acc / l ; LSE = m + log(l)
#pragma unroll
    for (int t = 0; t < 8; ++t) {
#pragma unroll
        for (int r = 0; r < 8; ++r) {
            const int row = qbase + half * 8 + r;
            Out[(size_t)row * HEAD_D + t * 16 + l16] = (_Float16)(o_acc[t][r] / lrow[r]);
        }
    }
    if (l16 == 0) {
#pragma unroll
        for (int r = 0; r < 8; ++r) {
            const int row = qbase + half * 8 + r;
            Lse[row] = (_Float16)(mrow[r] + __logf(lrow[r]));
        }
    }
}

extern "C" void kernel_launch(void* const* d_in, const int* in_sizes, int n_in,
                              void* d_out, int out_size, void* d_ws, size_t ws_size,
                              hipStream_t stream) {
    const _Float16* q = (const _Float16*)d_in[0];
    const _Float16* k = (const _Float16*)d_in[1];
    const _Float16* v = (const _Float16*)d_in[2];
    _Float16* out = (_Float16*)d_out;
    _Float16* o   = out;                                  // [8192,128] fp16
    _Float16* lse = out + (size_t)SEQ_LEN * HEAD_D;       // [8192] fp16

    dim3 grid(SEQ_LEN / BM);   // 64 workgroups
    dim3 block(256);           // 8 wave32s
    fa2_fwd_wmma_gfx1250<<<grid, block, 0, stream>>>(q, k, v, o, lse);
}
